// MultiHeadAttention_34892314312807
// MI455X (gfx1250) — compile-verified, matrix operand order checked by reading, not run on hardware
//
#include <hip/hip_runtime.h>

#define EMBED   2048
#define NHEAD   16
#define HDIM    128
#define BATCH   2
#define SEQ     2048
#define NROWS   (BATCH * SEQ)      // 4096
#define QKV_N   (3 * EMBED)        // 6144

typedef __attribute__((ext_vector_type(16))) __bf16 v16bf;
typedef __attribute__((ext_vector_type(8)))  float  v8f;

// hardware f32 -> bf16 convert (RNE on CDNA5)
__device__ __forceinline__ unsigned short f32_to_bf16_bits(float f) {
  __bf16 h = (__bf16)f;
  return __builtin_bit_cast(unsigned short, h);
}

// A/B 16-bit fragment (ISA 7.12.2): lanes 0-15 hold K {0..7,16..23},
// lanes 16-31 hold K {8..15,24..31} -> two contiguous 8-element runs.
union BfFrag { uint4 q[2]; v16bf v; };
__device__ __forceinline__ v16bf load_frag(const unsigned short* row, int kh) {
  BfFrag f;
  f.q[0] = *(const uint4*)(row + (kh << 3));        // elems 0..7  : k =  8*kh + 0..7
  f.q[1] = *(const uint4*)(row + 16 + (kh << 3));   // elems 8..15 : k = 16 + 8*kh + 0..7
  return f.v;
}

// pack 8 f32 -> 8 bf16 (one uint4)
__device__ __forceinline__ uint4 pack8v(float4 a, float4 b) {
  union { unsigned short u[8]; uint4 q; } o;
  o.u[0] = f32_to_bf16_bits(a.x); o.u[1] = f32_to_bf16_bits(a.y);
  o.u[2] = f32_to_bf16_bits(a.z); o.u[3] = f32_to_bf16_bits(a.w);
  o.u[4] = f32_to_bf16_bits(b.x); o.u[5] = f32_to_bf16_bits(b.y);
  o.u[6] = f32_to_bf16_bits(b.z); o.u[7] = f32_to_bf16_bits(b.w);
  return o.q;
}

// ---------------------------------------------------------------------------
// Kernel 1: qkv = x @ wqkv^T + bqkv, fused RoPE on q/k, scatter to (B,H,S,D)
// ---------------------------------------------------------------------------
__global__ void __launch_bounds__(256)
qkv_rope_kernel(const float* __restrict__ x,
                const float* __restrict__ wqkv,
                const float* __restrict__ bqkv,
                unsigned short* __restrict__ qws,
                unsigned short* __restrict__ kws,
                unsigned short* __restrict__ vws)
{
  __shared__ __align__(16) unsigned short As[128][32];  // bf16 tiles
  __shared__ __align__(16) unsigned short Bs[128][32];

  const int tid  = threadIdx.x;
  const int lane = tid & 31;
  const int w    = tid >> 5;
  const int lrow = lane & 15;
  const int kh   = lane >> 4;
  const int mBase = blockIdx.y * 128;
  const int nBase = blockIdx.x * 128;

  v8f acc[8] = {};

  for (int k0 = 0; k0 < EMBED; k0 += 32) {
    __syncthreads();
    {   // stage + convert once: 256 threads x 16 elems per tile
      const int r = tid >> 1;
      const int c = (tid & 1) << 4;
      const float4* ga = (const float4*)(x    + (size_t)(mBase + r) * EMBED + k0 + c);
      const float4* gb = (const float4*)(wqkv + (size_t)(nBase + r) * EMBED + k0 + c);
      *(uint4*)(&As[r][c])     = pack8v(ga[0], ga[1]);
      *(uint4*)(&As[r][c + 8]) = pack8v(ga[2], ga[3]);
      *(uint4*)(&Bs[r][c])     = pack8v(gb[0], gb[1]);
      *(uint4*)(&Bs[r][c + 8]) = pack8v(gb[2], gb[3]);
    }
    if (k0 + 32 < EMBED) {
      __builtin_prefetch(x    + (size_t)(mBase + (tid >> 1)) * EMBED + k0 + 32, 0, 1);
      __builtin_prefetch(wqkv + (size_t)(nBase + (tid >> 1)) * EMBED + k0 + 32, 0, 1);
    }
    __syncthreads();

    const v16bf a = load_frag(&As[w * 16 + lrow][0], kh);
#pragma unroll
    for (int f = 0; f < 8; ++f) {
      const v16bf bfrag = load_frag(&Bs[f * 16 + lrow][0], kh);
      acc[f] = __builtin_amdgcn_wmma_f32_16x16x32_bf16(
                 false, a, false, bfrag, (short)0, acc[f], false, false);
    }
  }

  // Each 128-wide N tile is exactly one (section, head) slab: n = sect*2048 + h*128 + d
  const int sect = nBase / EMBED;          // 0=q, 1=k, 2=v
  const int h    = (nBase % EMBED) / HDIM;

  if (sect < 2) {
    unsigned short* dst = (sect == 0) ? qws : kws;
    const float mul = (sect == 0) ? 0.08838834764831845f : 1.0f;  // fold D^-0.5 into q
#pragma unroll
    for (int f = 0; f < 4; ++f) {
      const int d64 = f * 16 + lrow;                    // pairs with d64+64 (frag f+4)
      const float invf = __powf(10000.0f, -(float)d64 * (1.0f / 64.0f));
      const float b0 = bqkv[nBase + d64];
      const float b1 = bqkv[nBase + d64 + 64];
#pragma unroll
      for (int r = 0; r < 8; ++r) {
        const int row = mBase + w * 16 + r + (kh << 3); // C layout: M = r + 8*half
        const int b = row >> 11;
        const int s = row & (SEQ - 1);
        float sn, c;
        __sincosf((float)s * invf, &sn, &c);
        const float x1 = acc[f][r]     + b0;
        const float x2 = acc[f + 4][r] + b1;
        const float o1 = (x1 * c - x2 * sn) * mul;
        const float o2 = (x2 * c + x1 * sn) * mul;
        const size_t base = (((size_t)(b * NHEAD + h)) * SEQ + s) * HDIM;
        dst[base + d64]      = f32_to_bf16_bits(o1);
        dst[base + d64 + 64] = f32_to_bf16_bits(o2);
      }
    }
  } else {
#pragma unroll
    for (int f = 0; f < 8; ++f) {
      const int d = f * 16 + lrow;
      const float bias = bqkv[nBase + d];
#pragma unroll
      for (int r = 0; r < 8; ++r) {
        const int row = mBase + w * 16 + r + (kh << 3);
        const int b = row >> 11;
        const int s = row & (SEQ - 1);
        vws[(((size_t)(b * NHEAD + h)) * SEQ + s) * HDIM + d] =
            f32_to_bf16_bits(acc[f][r] + bias);
      }
    }
  }
}

// ---------------------------------------------------------------------------
// Kernel 2: causal flash attention per (b, h, 128-row q tile)
// ---------------------------------------------------------------------------
__global__ void __launch_bounds__(256)
attn_kernel(const unsigned short* __restrict__ qws,
            const unsigned short* __restrict__ kws,
            const unsigned short* __restrict__ vws,
            float* __restrict__ ctx)
{
  __shared__ __align__(16) unsigned short Ks[128][128];  // K tile; reused as P tile
  __shared__ __align__(16) unsigned short Vt[128][128];  // V transposed: Vt[d][n]

  const int tid  = threadIdx.x;
  const int lane = tid & 31;
  const int w    = tid >> 5;
  const int lrow = lane & 15;
  const int kh   = lane >> 4;
  const int qt = blockIdx.x;
  const int h  = blockIdx.y;
  const int b  = blockIdx.z;
  const size_t headOff = ((size_t)(b * NHEAD + h)) * SEQ * HDIM;
  const int q0 = qt * 128;

  // Q A-fragments (16 q rows x 128 d) held in registers for the whole block
  v16bf aQ[4];
  {
    const unsigned short* qrow = qws + headOff + (size_t)(q0 + w * 16 + lrow) * HDIM;
#pragma unroll
    for (int kf = 0; kf < 4; ++kf) aQ[kf] = load_frag(qrow + kf * 32, kh);
  }

  v8f accO[8] = {};
  float m_i[8], l_i[8];
#pragma unroll
  for (int r = 0; r < 8; ++r) { m_i[r] = -1e30f; l_i[r] = 0.0f; }

  for (int t = 0; t <= qt; ++t) {
    const int kv0 = t * 128;
    __syncthreads();                       // all waves done with Ks(P)/Vt of prev tile
#pragma unroll
    for (int i = 0; i < 8; ++i) {
      int idx = tid + i * 256;             // 0..2047 chunks of 8 bf16
      int n   = idx >> 4;
      int c8  = (idx & 15) << 3;
      *(uint4*)(&Ks[n][c8]) = *(const uint4*)(kws + headOff + (size_t)(kv0 + n) * HDIM + c8);
      uint4 vv = *(const uint4*)(vws + headOff + (size_t)(kv0 + n) * HDIM + c8);
      const unsigned short* vs = (const unsigned short*)&vv;
#pragma unroll
      for (int j = 0; j < 8; ++j) Vt[c8 + j][n] = vs[j];
    }
    __syncthreads();

    // S = Q K^T : B[d][n] = K[n][d] -> contiguous ds_load_b128 pairs
    v8f sAcc[8] = {};
#pragma unroll
    for (int f = 0; f < 8; ++f) {
      const unsigned short* kp = &Ks[f * 16 + lrow][0];
#pragma unroll
      for (int kf = 0; kf < 4; ++kf) {
        const v16bf bk = load_frag(kp + kf * 32, kh);
        sAcc[f] = __builtin_amdgcn_wmma_f32_16x16x32_bf16(
                    false, aQ[kf], false, bk, (short)0, sAcc[f], false, false);
      }
    }
    __syncthreads();                       // everyone done reading Ks -> reuse as P

    // causal mask + online softmax (row = r + 8*kh, cols spread over 16 lanes x 8 frags)
    const bool diag = (t == qt);
    float newm[8];
#pragma unroll
    for (int r = 0; r < 8; ++r) newm[r] = m_i[r];
#pragma unroll
    for (int f = 0; f < 8; ++f) {
      const int kcol = kv0 + f * 16 + lrow;
#pragma unroll
      for (int r = 0; r < 8; ++r) {
        if (diag) {
          const int qrow = q0 + w * 16 + r + (kh << 3);
          if (kcol > qrow) sAcc[f][r] = -1e30f;
        }
        newm[r] = fmaxf(newm[r], sAcc[f][r]);
      }
    }
#pragma unroll
    for (int r = 0; r < 8; ++r) {          // xor masks < 16 keep the two row-halves separate
      float v = newm[r];
      v = fmaxf(v, __shfl_xor(v, 1));
      v = fmaxf(v, __shfl_xor(v, 2));
      v = fmaxf(v, __shfl_xor(v, 4));
      v = fmaxf(v, __shfl_xor(v, 8));
      newm[r] = v;
    }
    float corr[8], lsum[8];
#pragma unroll
    for (int r = 0; r < 8; ++r) {
      corr[r] = __expf(m_i[r] - newm[r]);
      m_i[r]  = newm[r];
      l_i[r] *= corr[r];
      lsum[r] = 0.0f;
    }
#pragma unroll
    for (int f = 0; f < 8; ++f)
#pragma unroll
      for (int r = 0; r < 8; ++r) {
        const float p = __expf(sAcc[f][r] - m_i[r]);
        sAcc[f][r] = p;
        lsum[r]   += p;
        accO[f][r] *= corr[r];
      }
#pragma unroll
    for (int r = 0; r < 8; ++r) {
      float v = lsum[r];
      v += __shfl_xor(v, 1);
      v += __shfl_xor(v, 2);
      v += __shfl_xor(v, 4);
      v += __shfl_xor(v, 8);
      l_i[r] += v;
    }

    // spill P (bf16) into this wave's private 16 rows of the Ks region
#pragma unroll
    for (int f = 0; f < 8; ++f) {
      const int n = f * 16 + lrow;
#pragma unroll
      for (int r = 0; r < 8; ++r)
        Ks[w * 16 + r + (kh << 3)][n] = f32_to_bf16_bits(sAcc[f][r]);
    }
    asm volatile("s_wait_dscnt 0" ::: "memory");   // LDS writes visible within wave

    // O += P V : A = P (rows = q), B[n][d] = Vt[d][n] -> contiguous reads
#pragma unroll
    for (int kf = 0; kf < 4; ++kf) {
      const v16bf aP = load_frag(&Ks[w * 16 + lrow][kf * 32], kh);
#pragma unroll
      for (int f = 0; f < 8; ++f) {
        const v16bf bv = load_frag(&Vt[f * 16 + lrow][kf * 32], kh);
        accO[f] = __builtin_amdgcn_wmma_f32_16x16x32_bf16(
                    false, aP, false, bv, (short)0, accO[f], false, false);
      }
    }
  }

  float invl[8];
#pragma unroll
  for (int r = 0; r < 8; ++r) invl[r] = 1.0f / l_i[r];
#pragma unroll
  for (int f = 0; f < 8; ++f) {
    const int d = f * 16 + lrow;
#pragma unroll
    for (int r = 0; r < 8; ++r) {
      const int s = q0 + w * 16 + r + (kh << 3);
      ctx[((size_t)(b * SEQ + s)) * EMBED + h * HDIM + d] = accO[f][r] * invl[r];
    }
  }
}

// ---------------------------------------------------------------------------
// Kernel 3: out = ctx @ wout^T + bout
// ---------------------------------------------------------------------------
__global__ void __launch_bounds__(256)
out_proj_kernel(const float* __restrict__ ctx,
                const float* __restrict__ wout,
                const float* __restrict__ bout,
                float* __restrict__ out)
{
  __shared__ __align__(16) unsigned short As[128][32];
  __shared__ __align__(16) unsigned short Bs[128][32];

  const int tid  = threadIdx.x;
  const int lane = tid & 31;
  const int w    = tid >> 5;
  const int lrow = lane & 15;
  const int kh   = lane >> 4;
  const int mBase = blockIdx.y * 128;
  const int nBase = blockIdx.x * 128;

  v8f acc[8] = {};

  for (int k0 = 0; k0 < EMBED; k0 += 32) {
    __syncthreads();
    {
      const int r = tid >> 1;
      const int c = (tid & 1) << 4;
      const float4* ga = (const float4*)(ctx  + (size_t)(mBase + r) * EMBED + k0 + c);
      const float4* gb = (const float4*)(wout + (size_t)(nBase + r) * EMBED + k0 + c);
      *(uint4*)(&As[r][c])     = pack8v(ga[0], ga[1]);
      *(uint4*)(&As[r][c + 8]) = pack8v(ga[2], ga[3]);
      *(uint4*)(&Bs[r][c])     = pack8v(gb[0], gb[1]);
      *(uint4*)(&Bs[r][c + 8]) = pack8v(gb[2], gb[3]);
    }
    if (k0 + 32 < EMBED) {
      __builtin_prefetch(ctx  + (size_t)(mBase + (tid >> 1)) * EMBED + k0 + 32, 0, 1);
      __builtin_prefetch(wout + (size_t)(nBase + (tid >> 1)) * EMBED + k0 + 32, 0, 1);
    }
    __syncthreads();

    const v16bf a = load_frag(&As[w * 16 + lrow][0], kh);
#pragma unroll
    for (int f = 0; f < 8; ++f) {
      const v16bf bfrag = load_frag(&Bs[f * 16 + lrow][0], kh);
      acc[f] = __builtin_amdgcn_wmma_f32_16x16x32_bf16(
                 false, a, false, bfrag, (short)0, acc[f], false, false);
    }
  }

#pragma unroll
  for (int f = 0; f < 8; ++f) {
    const int col = nBase + f * 16 + lrow;
    const float bias = bout[col];
#pragma unroll
    for (int r = 0; r < 8; ++r) {
      const int row = mBase + w * 16 + r + (kh << 3);
      out[(size_t)row * EMBED + col] = acc[f][r] + bias;
    }
  }
}

// ---------------------------------------------------------------------------
extern "C" void kernel_launch(void* const* d_in, const int* in_sizes, int n_in,
                              void* d_out, int out_size, void* d_ws, size_t ws_size,
                              hipStream_t stream) {
  (void)in_sizes; (void)n_in; (void)out_size; (void)ws_size;
  const float* x    = (const float*)d_in[0];
  // d_in[1] = causal mask (bool) -- handled analytically, unused
  const float* wqkv = (const float*)d_in[2];
  const float* bqkv = (const float*)d_in[3];
  const float* wout = (const float*)d_in[4];
  const float* bout = (const float*)d_in[5];
  float* out = (float*)d_out;

  const size_t headBytes = (size_t)BATCH * NHEAD * SEQ * HDIM * sizeof(unsigned short); // 16 MiB
  char* ws = (char*)d_ws;
  unsigned short* qws = (unsigned short*)(ws);
  unsigned short* kws = (unsigned short*)(ws + headBytes);
  unsigned short* vws = (unsigned short*)(ws + 2 * headBytes);
  float* ctx = (float*)(ws + 3 * headBytes);                    // 32 MiB f32

  qkv_rope_kernel<<<dim3(QKV_N / 128, NROWS / 128), 256, 0, stream>>>(
      x, wqkv, bqkv, qws, kws, vws);
  attn_kernel<<<dim3(SEQ / 128, NHEAD, BATCH), 256, 0, stream>>>(
      qws, kws, vws, ctx);
  out_proj_kernel<<<dim3(EMBED / 128, NROWS / 128), 256, 0, stream>>>(
      ctx, wout, bout, out);
}
